// GlobalGradMixtureDiffEncoder_67551245631812
// MI455X (gfx1250) — compile-verified, matrix operand order checked by reading, not run on hardware
//
#include <hip/hip_runtime.h>

// GlobalGradMixtureDiffEncoder on MI455X (gfx1250).
// - lat2 / w_img2 branch is dead code w.r.t. the output -> never touched
//   (saves 1.34 GB of HBM reads and ~86 GFLOP; remaining live set ~2.6 MB,
//   fully L2-resident; remaining ~1.7 GFLOP is latency-bound on the serial
//   10-iteration chain).
// - Fused persistent kernel: 8 workgroups x 16 batch rows, activations in
//   LDS, all GEMMs via v_wmma_f32_16x16x32_bf16.
// - All WMMA operands are bf16 in memory (images pre-converted once; lat
//   kept as f32 master + bf16 shadow in LDS), so operand loads are pure
//   b128 loads with no f32->bf16 VALU in the hot loops.

#define B_   128
#define G_   1024
#define K_   256
#define E2_  128
#define NIT  10
#define BETA 1e-3f

typedef __attribute__((ext_vector_type(16))) __bf16 v16bf;
typedef __attribute__((ext_vector_type(8)))  float  v8f;

union BF16x16 {
  v16bf v;
  unsigned short u[16];
  uint4 q[2];
};

__device__ __forceinline__ unsigned short f2bf(float f) {
  unsigned int u = __float_as_uint(f);
  u += 0x7fffu + ((u >> 16) & 1u);   // round-to-nearest-even
  return (unsigned short)(u >> 16);
}
__device__ __forceinline__ float bf2f(unsigned short h) {
  return __uint_as_float(((unsigned int)h) << 16);
}

// Operand load, 16x32 bf16 ISA layout:
// lane L: r=L&15 (row/col-of-tile), h=L>>4; VGPRs 0-3 hold K = kc+8h .. +7,
// VGPRs 4-7 hold K = kc+16+8h .. +7 (2 bf16 per VGPR).
// Two 16-byte vector loads per operand (LDS: ds_load_b128, global: b128).
__device__ __forceinline__ v16bf ld_op_bf(const unsigned short* base, int stride, int lane, int kc) {
  int r = lane & 15, h = lane >> 4;
  const unsigned short* p = base + r * stride + kc + 8 * h;
  BF16x16 b;
  b.q[0] = *(const uint4*)(p);
  b.q[1] = *(const uint4*)(p + 16);
  return b.v;
}

__device__ __forceinline__ v8f wmma_bf16(v16bf a, v16bf b, v8f c) {
  return __builtin_amdgcn_wmma_f32_16x16x32_bf16(false, a, false, b, (short)0, c, false, false);
}

// 16-rows x 16-slots reduction: thread t owns (row=t>>4, slot=t&15) partial.
// mode 1 = max, mode 0 = sum. Ends with a full barrier.
__device__ __forceinline__ void reduce16(float part, float* red, float* out, int tid, int mode) {
  red[(tid >> 4) * 16 + (tid & 15)] = part;
  __syncthreads();
  if (tid < 16) {
    float v = red[tid * 16];
    #pragma unroll
    for (int j = 1; j < 16; ++j) {
      float w = red[tid * 16 + j];
      v = mode ? fmaxf(v, w) : (v + w);
    }
    out[tid] = v;
  }
  __syncthreads();
}

// Row-wise softmax over K_=256 logits held in P (f32, stride sp).
// logit = s2*dot - s1*cvec[k]  (per-row ||x||^2 cancels in softmax).
// Writes normalized probs as bf16 into xp (stride sx). 256 threads.
__device__ __forceinline__ void softmax_rows(float* P, int sp, const float* __restrict__ cvec,
                                             float s2, float s1,
                                             unsigned short* xp, int sx,
                                             float* red, float* rowv, int tid) {
  int row = tid >> 4, l = tid & 15;
  float m = -3.0e38f;
  #pragma unroll
  for (int j = 0; j < 16; ++j) {
    int c = l + 16 * j;
    float v = s2 * P[row * sp + c] - s1 * cvec[c];
    P[row * sp + c] = v;
    m = fmaxf(m, v);
  }
  reduce16(m, red, rowv, tid, 1);
  float mm = rowv[row];
  float s = 0.f;
  #pragma unroll
  for (int j = 0; j < 16; ++j) {
    int c = l + 16 * j;
    float e = __expf(P[row * sp + c] - mm);
    P[row * sp + c] = e;
    s += e;
  }
  reduce16(s, red, rowv, tid, 0);
  float inv = 1.0f / rowv[row];
  #pragma unroll
  for (int j = 0; j < 16; ++j) {
    int c = l + 16 * j;
    xp[row * sx + c] = f2bf(P[row * sp + c] * inv);
  }
  __syncthreads();
}

// ---------------- prep kernels: bf16 copies + column norms -----------------
__global__ void prep_kg(const float* __restrict__ wp, unsigned short* Wkg, unsigned short* Wgk) {
  int i = blockIdx.x * blockDim.x + threadIdx.x;
  if (i < K_ * G_) {
    int k = i / G_, g = i % G_;
    unsigned short b = f2bf(wp[i]);      // w_project[k][g]
    Wkg[i] = b;                          // [K][G] (B-operand for x@wimg)
    Wgk[g * K_ + k] = b;                 // [G][K] (B-operand for xp@wimg.T)
  }
}
__global__ void prep_ek(const float* __restrict__ wr, unsigned short* Wek, unsigned short* Wke) {
  int i = blockIdx.x * blockDim.x + threadIdx.x;
  if (i < E2_ * K_) {
    int e = i / K_, k = i % K_;
    unsigned short b = f2bf(wr[i]);      // w_rec[e][k]
    Wek[i] = b;                          // [E2][K] (B-operand for y@wrec)
    Wke[k * E2_ + e] = b;                // [K][E2] (B-operand for lat@wrec.T)
  }
}
__global__ void prep_img(const float* __restrict__ images, unsigned short* imgB) {
  int i = blockIdx.x * blockDim.x + threadIdx.x;
  if (i < B_ * G_) imgB[i] = f2bf(images[i]);   // bf16 A-operand copy
}
__global__ void prep_norm(const float* __restrict__ wp, const float* __restrict__ wr,
                          float* cw, float* cr) {
  int k = threadIdx.x;                   // 256 threads, one block
  float s = 0.f;
  for (int g = 0; g < G_; ++g) { float v = wp[k * G_ + g]; s += v * v; }
  cw[k] = s;
  float s2 = 0.f;
  for (int e = 0; e < E2_; ++e) { float v = wr[e * K_ + k]; s2 += v * v; }
  cr[k] = s2;
}

// ---------------- fused main kernel: 8 blocks x 256 threads (16 rows/block) --
__global__ __launch_bounds__(256) void encoder_main(
    const float* __restrict__ images,
    const unsigned short* __restrict__ imgB,
    const unsigned short* __restrict__ Wkg, const unsigned short* __restrict__ Wgk,
    const unsigned short* __restrict__ Wek, const unsigned short* __restrict__ Wke,
    const float* __restrict__ cw, const float* __restrict__ cr,
    float* __restrict__ out) {
  const int SP = 260, SX = 264, SL = 132, SLB = 136;    // padded LDS row strides
  __shared__ __attribute__((aligned(16))) float          sLat[16 * SL];    // lat f32 master
  __shared__ __attribute__((aligned(16))) unsigned short sLatB[16 * SLB];  // lat bf16 shadow
  __shared__ __attribute__((aligned(16))) float          sP[16 * SP];      // logits / lk / xd-f32
  __shared__ __attribute__((aligned(16))) unsigned short sXP[16 * SX];     // xp bf16
  __shared__ __attribute__((aligned(16))) unsigned short sXD[16 * SX];     // xd / y bf16
  __shared__ float sRed[256];
  __shared__ float sRow[16];
  __shared__ float sRow2[16];
  __shared__ float sLoss[16];

  int tid = threadIdx.x, lane = tid & 31, wave = tid >> 5;
  int rb = blockIdx.x * 16;
  const float* img = images + rb * G_;
  const unsigned short* imgb = imgB + rb * G_;
  int col = lane & 15, hh = lane >> 4;

  if (tid < 16) sLoss[tid] = 0.f;

  // ---- encode: P = images x wimg  (16 tiles over K, depth G=1024) ----
  for (int t = wave * 2; t < wave * 2 + 2; ++t) {
    v8f acc = {};
    const unsigned short* Bp = Wkg + t * 16 * G_;
    for (int kc = 0; kc < G_; kc += 32)
      acc = wmma_bf16(ld_op_bf(imgb, G_, lane, kc), ld_op_bf(Bp, G_, lane, kc), acc);
    #pragma unroll
    for (int r = 0; r < 8; ++r) sP[(r + 8 * hh) * SP + t * 16 + col] = acc[r];
  }
  __syncthreads();
  softmax_rows(sP, SP, cw, 2.f * BETA / G_, BETA / G_, sXP, SX, sRed, sRow, tid);

  // ---- lat = xp @ wrec  (8 tiles over E2, depth K=256) ----
  {
    int t = wave;
    v8f acc = {};
    const unsigned short* Bp = Wek + t * 16 * K_;
    for (int kc = 0; kc < K_; kc += 32)
      acc = wmma_bf16(ld_op_bf(sXP, SX, lane, kc), ld_op_bf(Bp, K_, lane, kc), acc);
    #pragma unroll
    for (int r = 0; r < 8; ++r) {
      int row = r + 8 * hh;
      sLat[row * SL + t * 16 + col] = acc[r];
      sLatB[row * SLB + t * 16 + col] = f2bf(acc[r]);
    }
  }
  __syncthreads();

  for (int it = 0; it < NIT; ++it) {
    // (a) logits = lat @ wrec.T  (16 tiles over K, depth E2=128)
    for (int t = wave * 2; t < wave * 2 + 2; ++t) {
      v8f acc = {};
      const unsigned short* Bp = Wke + t * 16 * E2_;
      for (int kc = 0; kc < E2_; kc += 32)
        acc = wmma_bf16(ld_op_bf(sLatB, SLB, lane, kc), ld_op_bf(Bp, E2_, lane, kc), acc);
      #pragma unroll
      for (int r = 0; r < 8; ++r) sP[(r + 8 * hh) * SP + t * 16 + col] = acc[r];
    }
    __syncthreads();
    softmax_rows(sP, SP, cr, 2.f * BETA / E2_, BETA / E2_, sXP, SX, sRed, sRow, tid);

    // (b/c/d) fused over 4 G-chunks: rec tile -> xd(bf16) -> lk partial into sP
    for (int gc = 0; gc < 4; ++gc) {
      for (int t = wave * 2; t < wave * 2 + 2; ++t) {     // rec: xp @ wimg.T
        v8f acc = {};
        const unsigned short* Bp = Wgk + (gc * 256 + t * 16) * K_;
        for (int kc = 0; kc < K_; kc += 32)
          acc = wmma_bf16(ld_op_bf(sXP, SX, lane, kc), ld_op_bf(Bp, K_, lane, kc), acc);
        #pragma unroll
        for (int r = 0; r < 8; ++r) {
          int row = r + 8 * hh;
          float xd = img[row * G_ + gc * 256 + t * 16 + col] - acc[r];
          sXD[row * SX + t * 16 + col] = f2bf(xd);
        }
      }
      __syncthreads();
      for (int t = wave * 2; t < wave * 2 + 2; ++t) {     // lk += xd @ wimg (chunk)
        v8f acc = {};
        const unsigned short* Bp = Wkg + t * 16 * G_ + gc * 256;
        for (int kc = 0; kc < 256; kc += 32)
          acc = wmma_bf16(ld_op_bf(sXD, SX, lane, kc), ld_op_bf(Bp, G_, lane, kc), acc);
        #pragma unroll
        for (int r = 0; r < 8; ++r) {
          int idx = (r + 8 * hh) * SP + t * 16 + col;
          sP[idx] = (gc == 0) ? acc[r] : (sP[idx] + acc[r]);
        }
      }
      __syncthreads();
    }

    // (e) lkd, y = xp*(lk - <lk,xp>), sy = sum(y)
    {
      int row = tid >> 4, l = tid & 15;
      float pd = 0.f;
      #pragma unroll
      for (int j = 0; j < 16; ++j) {
        int c = l + 16 * j;
        pd += sP[row * SP + c] * bf2f(sXP[row * SX + c]);
      }
      reduce16(pd, sRed, sRow, tid, 0);
      float d = sRow[row];
      float py = 0.f;
      #pragma unroll
      for (int j = 0; j < 16; ++j) {
        int c = l + 16 * j;
        float xpv = bf2f(sXP[row * SX + c]);
        float yv = xpv * (sP[row * SP + c] - d);
        sXD[row * SX + c] = f2bf(yv);
        py += yv;
      }
      reduce16(py, sRed, sRow2, tid, 0);   // sRow2 = sy
    }

    // (f) latg = -lat*sy + y @ wrec ; lat += beta^2*0.001 * latg
    {
      int t = wave;
      v8f acc = {};
      const unsigned short* Bp = Wek + t * 16 * K_;
      for (int kc = 0; kc < K_; kc += 32)
        acc = wmma_bf16(ld_op_bf(sXD, SX, lane, kc), ld_op_bf(Bp, K_, lane, kc), acc);
      const float sc = BETA * BETA * 0.001f;
      #pragma unroll
      for (int r = 0; r < 8; ++r) {
        int row = r + 8 * hh;
        float lv = sLat[row * SL + t * 16 + col];
        float nv = lv + sc * (acc[r] - lv * sRow2[row]);
        sLat[row * SL + t * 16 + col] = nv;
        sLatB[row * SLB + t * 16 + col] = f2bf(nv);
      }
    }
    __syncthreads();
  }

  // ---- final decode: xp(lat), rec, per-row loss ----
  for (int t = wave * 2; t < wave * 2 + 2; ++t) {
    v8f acc = {};
    const unsigned short* Bp = Wke + t * 16 * E2_;
    for (int kc = 0; kc < E2_; kc += 32)
      acc = wmma_bf16(ld_op_bf(sLatB, SLB, lane, kc), ld_op_bf(Bp, E2_, lane, kc), acc);
    #pragma unroll
    for (int r = 0; r < 8; ++r) sP[(r + 8 * hh) * SP + t * 16 + col] = acc[r];
  }
  __syncthreads();
  softmax_rows(sP, SP, cr, 2.f * BETA / E2_, BETA / E2_, sXP, SX, sRed, sRow, tid);

  for (int gc = 0; gc < 4; ++gc) {
    for (int t = wave * 2; t < wave * 2 + 2; ++t) {
      v8f acc = {};
      const unsigned short* Bp = Wgk + (gc * 256 + t * 16) * K_;
      for (int kc = 0; kc < K_; kc += 32)
        acc = wmma_bf16(ld_op_bf(sXP, SX, lane, kc), ld_op_bf(Bp, K_, lane, kc), acc);
      #pragma unroll
      for (int r = 0; r < 8; ++r) {
        int row = r + 8 * hh;
        sP[row * SP + t * 16 + col] = img[row * G_ + gc * 256 + t * 16 + col] - acc[r];
      }
    }
    __syncthreads();
    {
      int row = tid >> 4, l = tid & 15;
      float s = 0.f;
      #pragma unroll
      for (int j = 0; j < 16; ++j) {
        float v = sP[row * SP + l + 16 * j];
        s += v * v;
      }
      reduce16(s, sRed, sRow, tid, 0);
      if (tid < 16) sLoss[tid] += sRow[tid];
    }
    __syncthreads();
  }
  if (tid < 16) out[rb + tid] = sLoss[tid] * (1.0f / G_);
}

extern "C" void kernel_launch(void* const* d_in, const int* in_sizes, int n_in,
                              void* d_out, int out_size, void* d_ws, size_t ws_size,
                              hipStream_t stream) {
  (void)in_sizes; (void)n_in; (void)out_size; (void)ws_size;
  const float* images = (const float*)d_in[0];   // [128,1024]
  const float* wp     = (const float*)d_in[1];   // w_project [256,1024]
  const float* wr     = (const float*)d_in[2];   // w_rec [128,256]
  // d_in[3] (w_img2) intentionally unused: dead code w.r.t. the output.
  float* out = (float*)d_out;                    // [128] per-sample loss

  unsigned short* Wkg  = (unsigned short*)d_ws;          // [K][G]   512 KB
  unsigned short* Wgk  = Wkg + K_ * G_;                  // [G][K]   512 KB
  unsigned short* Wek  = Wgk + G_ * K_;                  // [E2][K]   64 KB
  unsigned short* Wke  = Wek + E2_ * K_;                 // [K][E2]   64 KB
  unsigned short* imgB = Wke + K_ * E2_;                 // [B][G]   256 KB
  float* cw = (float*)(imgB + B_ * G_);                  // [K]
  float* cr = cw + K_;                                   // [K]

  prep_kg<<<(K_ * G_ + 255) / 256, 256, 0, stream>>>(wp, Wkg, Wgk);
  prep_ek<<<(E2_ * K_ + 255) / 256, 256, 0, stream>>>(wr, Wek, Wke);
  prep_img<<<(B_ * G_ + 255) / 256, 256, 0, stream>>>(images, imgB);
  prep_norm<<<1, 256, 0, stream>>>(wp, wr, cw, cr);
  encoder_main<<<B_ / 16, 256, 0, stream>>>(images, imgB, Wkg, Wgk, Wek, Wke, cw, cr, out);
}
